// SpatialCrossAttention_80341658239610
// MI455X (gfx1250) — matrix operand order repaired and checked by reading, not compiled, not run on hardware
//
#include <hip/hip_runtime.h>

typedef __bf16 bf16_t;
typedef __attribute__((ext_vector_type(16))) __bf16 v16bf;
typedef __attribute__((ext_vector_type(8)))  __bf16 v8bf;
typedef __attribute__((ext_vector_type(8)))  float  v8f;

#define Cn   256
#define Dn   4096
#define HWn  4096
#define Bn   4

// ---------------------------------------------------------------------------
// WMMA helpers (CDNA5 wave32, 16x16x32 bf16 -> f32)
// ---------------------------------------------------------------------------
__device__ __forceinline__ v8f wmma_bf16(v16bf a, v16bf b, v8f c) {
  return __builtin_amdgcn_wmma_f32_16x16x32_bf16(false, a, false, b, (short)0, c,
                                                 false, false);
}

// A-matrix 16x32 fragment (MxK), source row-major [16+][ld] bf16.
// lane<16: M=lane, elems 0..15 -> K k0..k0+15 contiguous
// lane>=16: M=lane-16, K offset +16.
__device__ __forceinline__ v16bf load_a_frag(const bf16_t* base, int ld, int k0,
                                             int lane) {
  int m  = lane & 15;
  int kb = k0 + ((lane & 16) ? 16 : 0);
  const bf16_t* p = base + (size_t)m * ld + kb;
  v8bf lo = *(const v8bf*)(p);
  v8bf hi = *(const v8bf*)(p + 8);
  v16bf a;
#pragma unroll
  for (int e = 0; e < 8; ++e) { a[e] = lo[e]; a[8 + e] = hi[e]; }
  return a;
}

// B-matrix 32x16 fragment (KxN) from N-major storage [N][ldk]:
// lane<16: N=n0+lane, elems = K k0..k0+15 contiguous; lane>=16: K+16.
__device__ __forceinline__ v16bf load_b_frag(const bf16_t* nmajor, int n0, int ldk,
                                             int k0, int lane) {
  const bf16_t* p =
      nmajor + (size_t)(n0 + (lane & 15)) * ldk + k0 + ((lane & 16) ? 16 : 0);
  return *(const v16bf*)p;
}

// ---------------------------------------------------------------------------
// Weight prep: f32 -> bf16, transposed to N-major [N][K] where needed
// ---------------------------------------------------------------------------
__global__ void prep_weights_kernel(const float* qw, const float* w2, const float* w3,
                                    const float* w4, const float* pw, bf16_t* qwb,
                                    bf16_t* w2t, bf16_t* w3t, bf16_t* w4t,
                                    bf16_t* pwb) {
  int idx = blockIdx.x * 256 + threadIdx.x;
  if (idx < 65536) { qwb[idx] = (bf16_t)qw[idx]; return; }          // [o][c]=[N][K]
  idx -= 65536;
  if (idx < 32768) {                                                 // kv_w2 [128][256]
    int n = idx >> 7, k = idx & 127;
    w2t[idx] = (bf16_t)w2[k * 256 + n];
    return;
  }
  idx -= 32768;
  if (idx < 65536) {                                                 // kv_w3 [256][256]
    int n = idx >> 8, k = idx & 255;
    w3t[idx] = (bf16_t)w3[k * 256 + n];
    return;
  }
  idx -= 65536;
  if (idx < 131072) {                                                // kv_w4 [256][512]
    int n = idx >> 8, k = idx & 255;
    w4t[idx] = (bf16_t)w4[k * 512 + n];
    return;
  }
  idx -= 131072;
  if (idx < 65536) pwb[idx] = (bf16_t)pw[idx];                       // [o][c]=[M][K]
}

// y [b][C][hw] f32  ->  yT [b][hw][C] bf16
__global__ void y_transpose_kernel(const float* y, bf16_t* yT) {
  size_t idx = (size_t)blockIdx.x * 256 + threadIdx.x;
  if (idx >= (size_t)Bn * Cn * HWn) return;
  int i = (int)(idx & 4095);
  int c = (int)((idx >> 12) & 255);
  int b = (int)(idx >> 20);
  yT[(((size_t)b << 12) + i) * Cn + c] = (bf16_t)y[idx];
}

// ---------------------------------------------------------------------------
// kv generator MLP: one wave per block, TWO 16-row A-tiles per wave so each
// weight B-fragment is loaded once and feeds two WMMAs (2x weight reuse).
// ---------------------------------------------------------------------------
__device__ __forceinline__ void ln_leaky2(float* af, bf16_t* ab, float* mu_s,
                                          float* rs_s, int W, int logW,
                                          const float* g, const float* bb,
                                          int lane) {
  {  // 32 rows total (2 tiles x 16), one row per lane
    int tile = lane >> 4, r = lane & 15;
    const float* rowp = af + tile * 4096 + r * 256;
    float s = 0.f, s2 = 0.f;
    for (int c = 0; c < W; ++c) { float v = rowp[c]; s += v; s2 += v * v; }
    float mu  = s / (float)W;
    float var = s2 / (float)W - mu * mu;
    mu_s[lane] = mu;
    rs_s[lane] = __frsqrt_rn(var + 1e-5f);
  }
  for (int idx = lane; idx < (32 << logW); idx += 32) {
    int tile = idx >> (logW + 4);
    int rem  = idx & ((16 << logW) - 1);
    int r = rem >> logW, c = rem & (W - 1);
    float v = (af[tile * 4096 + r * 256 + c] - mu_s[tile * 16 + r]) *
                  rs_s[tile * 16 + r] * g[c] + bb[c];
    v = (v < 0.f) ? 0.01f * v : v;
    ab[tile * 4096 + r * 256 + c] = (bf16_t)v;
  }
}

__global__ __launch_bounds__(32) void kvgen_kernel(
    const float* x, const float* w1, const float* b1, const float* g1,
    const float* bb1, const bf16_t* w2t, const float* b2, const float* g2,
    const float* bb2, const bf16_t* w3t, const float* b3, const float* g3,
    const float* bb3, const bf16_t* w4t, const float* b4, bf16_t* Kt, bf16_t* Vt) {
  __shared__ float  actf[2][16 * 256];
  __shared__ bf16_t actb[2][16 * 256];
  __shared__ float  mu_s[32], rs_s[32], xs[32];

  int lane = threadIdx.x;
  int row0 = blockIdx.x * 32;  // global row over b*D; tiles at row0, row0+16
  float*  af = &actf[0][0];
  bf16_t* ab = &actb[0][0];

  xs[lane] = x[row0 + lane];

  // layer1: scalar -> 128 (both tiles)
  for (int idx = lane; idx < 4096; idx += 32) {
    int tile = idx >> 11, r = (idx >> 7) & 15, c = idx & 127;
    actf[tile][r * 256 + c] = xs[idx >> 7] * w1[c] + b1[c];
  }
  ln_leaky2(af, ab, mu_s, rs_s, 128, 7, g1, bb1, lane);

  // layer2: [16x128] @ [128x256], x2 tiles
  for (int nt = 0; nt < 16; ++nt) {
    v8f a0 = {}, a1 = {};
#pragma unroll
    for (int ks = 0; ks < 4; ++ks) {
      v16bf bm = load_b_frag(w2t, nt * 16, 128, ks * 32, lane);
      a0 = wmma_bf16(load_a_frag(ab, 256, ks * 32, lane), bm, a0);
      a1 = wmma_bf16(load_a_frag(ab + 4096, 256, ks * 32, lane), bm, a1);
    }
#pragma unroll
    for (int r = 0; r < 8; ++r) {
      int m = r + ((lane & 16) ? 8 : 0), n = nt * 16 + (lane & 15);
      af[m * 256 + n]        = a0[r] + b2[n];
      af[4096 + m * 256 + n] = a1[r] + b2[n];
    }
  }
  ln_leaky2(af, ab, mu_s, rs_s, 256, 8, g2, bb2, lane);

  // layer3: [16x256] @ [256x256], x2 tiles
  for (int nt = 0; nt < 16; ++nt) {
    v8f a0 = {}, a1 = {};
#pragma unroll
    for (int ks = 0; ks < 8; ++ks) {
      v16bf bm = load_b_frag(w3t, nt * 16, 256, ks * 32, lane);
      a0 = wmma_bf16(load_a_frag(ab, 256, ks * 32, lane), bm, a0);
      a1 = wmma_bf16(load_a_frag(ab + 4096, 256, ks * 32, lane), bm, a1);
    }
#pragma unroll
    for (int r = 0; r < 8; ++r) {
      int m = r + ((lane & 16) ? 8 : 0), n = nt * 16 + (lane & 15);
      af[m * 256 + n]        = a0[r] + b3[n];
      af[4096 + m * 256 + n] = a1[r] + b3[n];
    }
  }
  ln_leaky2(af, ab, mu_s, rs_s, 256, 8, g3, bb3, lane);

  // layer4: [16x256] @ [256x512], silu, split k|v, x2 tiles
  for (int nt = 0; nt < 32; ++nt) {
    v8f a0 = {}, a1 = {};
#pragma unroll
    for (int ks = 0; ks < 8; ++ks) {
      v16bf bm = load_b_frag(w4t, nt * 16, 256, ks * 32, lane);
      a0 = wmma_bf16(load_a_frag(ab, 256, ks * 32, lane), bm, a0);
      a1 = wmma_bf16(load_a_frag(ab + 4096, 256, ks * 32, lane), bm, a1);
    }
#pragma unroll
    for (int r = 0; r < 8; ++r) {
      int m = r + ((lane & 16) ? 8 : 0), n = nt * 16 + (lane & 15);
#pragma unroll
      for (int t2 = 0; t2 < 2; ++t2) {
        float v = (t2 ? a1[r] : a0[r]) + b4[n];
        v = v * (1.f / (1.f + __expf(-v)));  // silu
        int rg = row0 + t2 * 16 + m;
        if (n < 256) {
          Kt[(size_t)rg * Cn + n] = (bf16_t)v;                        // [b*D][C]
        } else {
          int bidx = rg >> 12, dd = rg & 4095;
          Vt[(((size_t)bidx << 8) + (n - 256)) * Dn + dd] = (bf16_t)v;  // [b][C][D]
        }
      }
    }
  }
}

// ---------------------------------------------------------------------------
// q = dw-affine(conv1x1(y)) : [hw x C] @ qw^T, bf16 out
// ---------------------------------------------------------------------------
__global__ __launch_bounds__(256) void qgen_kernel(const bf16_t* yT,
                                                   const bf16_t* qwb,
                                                   const float* q_b,
                                                   const float* dw_w,
                                                   const float* dw_b, bf16_t* Qb) {
  int lane = threadIdx.x & 31, wv = threadIdx.x >> 5;
  int b = blockIdx.x >> 5;
  int i0 = (blockIdx.x & 31) * 128 + wv * 16;
  const bf16_t* arow = yT + ((size_t)b * HWn + i0) * Cn;

  v16bf qa[8];
#pragma unroll
  for (int ks = 0; ks < 8; ++ks) qa[ks] = load_a_frag(arow, Cn, ks * 32, lane);

  for (int nt = 0; nt < 16; ++nt) {
    v8f acc = {};
#pragma unroll
    for (int ks = 0; ks < 8; ++ks)
      acc = wmma_bf16(qa[ks], load_b_frag(qwb, nt * 16, Cn, ks * 32, lane), acc);
#pragma unroll
    for (int r = 0; r < 8; ++r) {
      int m = r + ((lane & 16) ? 8 : 0), n = nt * 16 + (lane & 15);
      float v = (acc[r] + q_b[n]) * dw_w[n] + dw_b[n];
      Qb[((size_t)b * HWn + i0 + m) * Cn + n] = (bf16_t)v;
    }
  }
}

// ---------------------------------------------------------------------------
// Fused flash attention. K/V tiles staged in LDS once per WG per iteration
// (8 waves share them -> 8x less L2 traffic); per-wave online softmax.
// ~116 KB LDS -> 2 workgroups per 320KB WGP.
// ---------------------------------------------------------------------------
__global__ __launch_bounds__(256) void attn_kernel(const bf16_t* Q, const bf16_t* Kt,
                                                   const bf16_t* Vt, bf16_t* Hb) {
  __shared__ __align__(32) bf16_t Ktile[64 * 256];   // [dd_local][c]   32KB
  __shared__ __align__(32) bf16_t Vtile[256 * 64];   // [c][dd_local]   32KB
  __shared__ float  sc[8][16][64];                   //                 32KB
  __shared__ bf16_t pb[8][16][64];                   //                 16KB
  __shared__ float  rmax[8][16], rsum[8][16], sfac[8][16];
  __shared__ float  pmax[8][32], psum[8][32];

  int tid  = threadIdx.x;
  int lane = tid & 31, wv = tid >> 5;
  int b  = blockIdx.x >> 5;
  int i0 = (blockIdx.x & 31) * 128 + wv * 16;
  const bf16_t* qb  = Q + ((size_t)b * HWn + i0) * Cn;
  const bf16_t* ktb = Kt + (size_t)b * Dn * Cn;   // [D][C] row-major (row=dd)
  const bf16_t* vtb = Vt + (size_t)b * Cn * Dn;   // [C][D] (row=c)

  v16bf qa[8];
#pragma unroll
  for (int ks = 0; ks < 8; ++ks) qa[ks] = load_a_frag(qb, Cn, ks * 32, lane);

  v8f oacc[16];
#pragma unroll
  for (int nt = 0; nt < 16; ++nt) oacc[nt] = (v8f){};
  if (lane < 16) { rmax[wv][lane] = -3.0e38f; rsum[wv][lane] = 0.f; }

  int row  = lane & 15;
  int half = (lane >> 4) * 32;  // softmax column half owned by this lane

  for (int kt = 0; kt < Dn / 64; ++kt) {
    int dd0 = kt * 64;
    __syncthreads();  // previous tile fully consumed by all waves
    {   // ---- stage K tile: 32KB contiguous ----
      const v8bf* ksrc = (const v8bf*)(ktb + (size_t)dd0 * Cn);
      v8bf* kdst = (v8bf*)Ktile;
      for (int j = tid; j < 2048; j += 256) kdst[j] = ksrc[j];
      // ---- stage V tile: one 128B row slice per thread ----
      const v8bf* vsrc = (const v8bf*)(vtb + (size_t)tid * Dn + dd0);
      v8bf* vdst = (v8bf*)(Vtile + tid * 64);
#pragma unroll
      for (int j = 0; j < 8; ++j) vdst[j] = vsrc[j];
    }
    __syncthreads();  // tiles visible to all waves

    // ---- scores: [16x256] @ K^T tile [256x64], *C^-0.5 ----
    for (int nt = 0; nt < 4; ++nt) {
      v8f s = {};
#pragma unroll
      for (int ks = 0; ks < 8; ++ks)
        s = wmma_bf16(qa[ks], load_b_frag(Ktile, nt * 16, Cn, ks * 32, lane), s);
#pragma unroll
      for (int r = 0; r < 8; ++r) {
        int rr = r + ((lane & 16) ? 8 : 0);
        sc[wv][rr][nt * 16 + (lane & 15)] = s[r] * 0.0625f;  // 1/sqrt(256)
      }
    }
    // ---- online softmax: 2 lanes per row ----
    float mx = -3.0e38f;
    for (int j = 0; j < 32; ++j) mx = fmaxf(mx, sc[wv][row][half + j]);
    pmax[wv][lane] = mx;
    float tmx  = fmaxf(pmax[wv][row], pmax[wv][row + 16]);
    float oldm = rmax[wv][row];
    float nm   = fmaxf(oldm, tmx);
    float sf   = __expf(oldm - nm);
    float sm = 0.f;
    for (int j = 0; j < 32; ++j) {
      float e = __expf(sc[wv][row][half + j] - nm);
      pb[wv][row][half + j] = (bf16_t)e;
      sm += e;
    }
    psum[wv][lane] = sm;
    if (lane < 16) {
      rsum[wv][row] = rsum[wv][row] * sf + psum[wv][row] + psum[wv][row + 16];
      rmax[wv][row] = nm;
      sfac[wv][row] = sf;
    }
    // ---- rescale running accumulator ----
    float fr[8];
#pragma unroll
    for (int r = 0; r < 8; ++r) fr[r] = sfac[wv][r + ((lane & 16) ? 8 : 0)];
    for (int nt = 0; nt < 16; ++nt)
#pragma unroll
      for (int r = 0; r < 8; ++r) oacc[nt][r] *= fr[r];
    // ---- O += P @ V tile : [16x64] @ [64x256] ----
#pragma unroll
    for (int ks2 = 0; ks2 < 2; ++ks2) {
      v16bf pa = load_a_frag(&pb[wv][0][0], 64, ks2 * 32, lane);
      for (int nt = 0; nt < 16; ++nt)
        oacc[nt] = wmma_bf16(
            pa, load_b_frag(Vtile, nt * 16, 64, ks2 * 32, lane), oacc[nt]);
    }
  }
  // ---- normalize + store bf16 ----
  float inv8[8];
#pragma unroll
  for (int r = 0; r < 8; ++r) inv8[r] = 1.f / rsum[wv][r + ((lane & 16) ? 8 : 0)];
  for (int nt = 0; nt < 16; ++nt)
#pragma unroll
    for (int r = 0; r < 8; ++r) {
      int m = r + ((lane & 16) ? 8 : 0), n = nt * 16 + (lane & 15);
      Hb[((size_t)b * HWn + i0 + m) * Cn + n] = (bf16_t)(oacc[nt][r] * inv8[r]);
    }
}

// ---------------------------------------------------------------------------
// out = y + proj(h) : A = proj_w [o][c], B = h [i][c], f32 residual store
// ---------------------------------------------------------------------------
__global__ __launch_bounds__(256) void proj_kernel(const bf16_t* Hb,
                                                   const bf16_t* pwb,
                                                   const float* proj_b,
                                                   const float* y, float* out) {
  int lane = threadIdx.x & 31, wv = threadIdx.x >> 5;
  int b = blockIdx.x >> 5;
  int i0 = (blockIdx.x & 31) * 128 + wv * 16;
  const bf16_t* hbase = Hb + (size_t)b * HWn * Cn;

  for (int mt = 0; mt < 16; ++mt) {
    v8f acc = {};
#pragma unroll
    for (int ks = 0; ks < 8; ++ks) {
      v16bf a  = load_a_frag(pwb + mt * 16 * Cn, Cn, ks * 32, lane);
      v16bf bb = load_b_frag(hbase, i0, Cn, ks * 32, lane);
      acc = wmma_bf16(a, bb, acc);
    }
#pragma unroll
    for (int r = 0; r < 8; ++r) {
      int o = mt * 16 + r + ((lane & 16) ? 8 : 0);
      int i = i0 + (lane & 15);
      size_t oi = ((size_t)b * Cn + o) * HWn + i;
      out[oi] = y[oi] + acc[r] + proj_b[o];
    }
  }
}

// ---------------------------------------------------------------------------
// launch
// ---------------------------------------------------------------------------
extern "C" void kernel_launch(void* const* d_in, const int* in_sizes, int n_in,
                              void* d_out, int out_size, void* d_ws, size_t ws_size,
                              hipStream_t stream) {
  (void)in_sizes; (void)n_in; (void)out_size; (void)ws_size;
  const float* x     = (const float*)d_in[0];
  const float* y     = (const float*)d_in[1];
  const float* q_w   = (const float*)d_in[2];
  const float* q_b   = (const float*)d_in[3];
  const float* dw_w  = (const float*)d_in[4];
  const float* dw_b  = (const float*)d_in[5];
  const float* kv_w1 = (const float*)d_in[6];
  const float* kv_b1 = (const float*)d_in[7];
  const float* ln1_g = (const float*)d_in[8];
  const float* ln1_b = (const float*)d_in[9];
  const float* kv_w2 = (const float*)d_in[10];
  const float* kv_b2 = (const float*)d_in[11];
  const float* ln2_g = (const float*)d_in[12];
  const float* ln2_b = (const float*)d_in[13];
  const float* kv_w3 = (const float*)d_in[14];
  const float* kv_b3 = (const float*)d_in[15];
  const float* ln3_g = (const float*)d_in[16];
  const float* ln3_b = (const float*)d_in[17];
  const float* kv_w4 = (const float*)d_in[18];
  const float* kv_b4 = (const float*)d_in[19];
  const float* pj_w  = (const float*)d_in[20];
  const float* pj_b  = (const float*)d_in[21];
  float* out = (float*)d_out;

  char* ws = (char*)d_ws;
  const size_t SZ = (size_t)8 * 1024 * 1024;  // each bf16 activation buffer
  bf16_t* yT  = (bf16_t*)(ws);
  bf16_t* Qb  = (bf16_t*)(ws + SZ);
  bf16_t* Kt  = (bf16_t*)(ws + 2 * SZ);
  bf16_t* Vt  = (bf16_t*)(ws + 3 * SZ);
  bf16_t* Hb  = (bf16_t*)(ws + 4 * SZ);
  char*   wp  = ws + 5 * SZ;
  bf16_t* qwb = (bf16_t*)(wp);             // 131072 B
  bf16_t* w2t = (bf16_t*)(wp + 131072);    //  65536 B
  bf16_t* w3t = (bf16_t*)(wp + 196608);    // 131072 B
  bf16_t* w4t = (bf16_t*)(wp + 327680);    // 262144 B
  bf16_t* pwb = (bf16_t*)(wp + 589824);    // 131072 B

  prep_weights_kernel<<<1408, 256, 0, stream>>>(q_w, kv_w2, kv_w3, kv_w4, pj_w,
                                                qwb, w2t, w3t, w4t, pwb);
  y_transpose_kernel<<<(Bn * Cn * HWn) / 256, 256, 0, stream>>>(y, yT);
  kvgen_kernel<<<(Bn * Dn) / 32, 32, 0, stream>>>(
      x, kv_w1, kv_b1, ln1_g, ln1_b, w2t, kv_b2, ln2_g, ln2_b, w3t, kv_b3, ln3_g,
      ln3_b, w4t, kv_b4, Kt, Vt);
  qgen_kernel<<<Bn * 32, 256, 0, stream>>>(yT, qwb, q_b, dw_w, dw_b, Qb);
  attn_kernel<<<Bn * 32, 256, 0, stream>>>(Qb, Kt, Vt, Hb);
  proj_kernel<<<Bn * 32, 256, 0, stream>>>(Hb, pwb, pj_b, y, out);
}
